// FPN2MLPFeatureExtractorARRM_18554258719319
// MI455X (gfx1250) — matrix operand order repaired and checked by reading, not compiled, not run on hardware
//
#include <hip/hip_runtime.h>
#include <hip/hip_bf16.h>

typedef _Float16 half_t;
typedef __attribute__((ext_vector_type(16))) _Float16 v16h;
typedef __attribute__((ext_vector_type(8)))  _Float16 v8h;
typedef __attribute__((ext_vector_type(8)))  float    v8f;

#define BM 128
#define BN 128
#define STR 40   // halves per LDS row: 32 data + 8 pad (keeps 16B alignment)

enum { OM_F32_RM = 0, OM_F16_RM = 1, OM_F32_NCHW = 2, OM_F16_NCHW = 3, OM_F16_TN = 4 };

// ---------------------------------------------------------------------------
// WMMA GEMM: out(M,N) = A(M,K) @ B^T  [+bias] [+acc] [relu]
//   A: f16 row-major M x K (lda)
//   B: f16 row-major N x K (ldb) -- caller guarantees >= ceil(N/128)*128 rows
// M % 128 == 0, K % 32 == 0. N guards applied on stores only.
// Double-buffered LDS staging via GLOBAL_LOAD_ASYNC_TO_LDS_B128 (ASYNCcnt).
// ---------------------------------------------------------------------------
__global__ __launch_bounds__(256) void wmma_gemm_kernel(
    const half_t* __restrict__ A, int lda,
    const half_t* __restrict__ B, int ldb,
    void* __restrict__ outp, int ldo, int outMode,
    const float* __restrict__ bias, int accumulate, int relu,
    int M, int N, int K, int rowOffset, int HW, int Cch) {
  // [ lA buf0 | lA buf1 | lB buf0 | lB buf1 ], each 128 x STR halves
  __shared__ __align__(16) half_t sm[4 * BM * STR];

  const int tid  = threadIdx.x;
  const int lane = tid & 31;
  const int wave = tid >> 5;
  const int wr   = wave >> 1;  // 0..3 : 32-row strip
  const int wc   = wave & 1;   // 0..1 : 64-col strip
  const int rowBase = blockIdx.y * BM;
  const int colBase = blockIdx.x * BN;
  const int m16 = lane & 15;
  const int g   = lane >> 4;

  // staging: thread -> (row sr & sr+64, 16B chunk sc) of the 128x32 tile
  const int sr = tid >> 2;
  const int sc = (tid & 3) * 8;
  const half_t* aS0 = A + (size_t)(rowBase + sr) * lda + sc;
  const half_t* aS1 = aS0 + (size_t)64 * lda;
  const half_t* bS0 = B + (size_t)(colBase + sr) * ldb + sc;
  const half_t* bS1 = bS0 + (size_t)64 * ldb;

  // ISA 10.2: generic-pointer low 32 bits == wave-relative LDS address
  const unsigned smBase = (unsigned)(unsigned long long)(const void*)sm;
  const unsigned aOff = smBase + (unsigned)(sr * STR + sc) * 2;
  const unsigned bOff = smBase + (unsigned)(2 * BM * STR + sr * STR + sc) * 2;
  const unsigned BUFB = (unsigned)BM * STR * 2;   // bytes per buffer
  const unsigned H64  = (unsigned)64 * STR * 2;   // second 64-row half

  auto issue = [&](int kb, int buf) {
    unsigned a = aOff + (unsigned)buf * BUFB;
    unsigned b = bOff + (unsigned)buf * BUFB;
    asm volatile("global_load_async_to_lds_b128 %0, %1, off"
                 :: "v"(a), "v"((unsigned long long)(aS0 + kb)) : "memory");
    asm volatile("global_load_async_to_lds_b128 %0, %1, off"
                 :: "v"(a + H64), "v"((unsigned long long)(aS1 + kb)) : "memory");
    asm volatile("global_load_async_to_lds_b128 %0, %1, off"
                 :: "v"(b), "v"((unsigned long long)(bS0 + kb)) : "memory");
    asm volatile("global_load_async_to_lds_b128 %0, %1, off"
                 :: "v"(b + H64), "v"((unsigned long long)(bS1 + kb)) : "memory");
  };

  v8f acc[2][4] = {};

  const int T = K / 32;
  issue(0, 0);
  for (int t = 0; t < T; ++t) {
    const int buf = t & 1;
    if (t + 1 < T) {
      issue((t + 1) * 32, buf ^ 1);
      asm volatile("s_wait_asynccnt 0x4" ::: "memory");  // current tile landed
    } else {
      asm volatile("s_wait_asynccnt 0x0" ::: "memory");
    }
    __syncthreads();

    const half_t* tA = sm + (size_t)buf * (BM * STR);
    const half_t* tB = sm + 2 * (BM * STR) + (size_t)buf * (BM * STR);

    // fragments per ISA 7.12.2 layouts
    v16h aF[2], bF[4];
#pragma unroll
    for (int i = 0; i < 2; ++i) {
      const half_t* p = tA + (wr * 32 + i * 16 + m16) * STR + g * 8;
      v8h lo = *(const v8h*)p;
      v8h hi = *(const v8h*)(p + 16);
      v16h f;
#pragma unroll
      for (int e = 0; e < 8; ++e) { f[e] = lo[e]; f[e + 8] = hi[e]; }
      aF[i] = f;
    }
#pragma unroll
    for (int j = 0; j < 4; ++j) {
      const half_t* p = tB + (wc * 64 + j * 16 + m16) * STR + g * 8;
      v8h lo = *(const v8h*)p;
      v8h hi = *(const v8h*)(p + 16);
      v16h f;
#pragma unroll
      for (int e = 0; e < 8; ++e) { f[e] = lo[e]; f[e + 8] = hi[e]; }
      bF[j] = f;
    }
#pragma unroll
    for (int i = 0; i < 2; ++i)
#pragma unroll
      for (int j = 0; j < 4; ++j)
        acc[i][j] = __builtin_amdgcn_wmma_f32_16x16x32_f16(
            false, aF[i], false, bF[j], (short)0, acc[i][j], false, false);
    __syncthreads();
  }

  // store: C/D layout = lane(0-15) N=lane, vgpr v -> M=v (+8 for lanes 16-31)
#pragma unroll
  for (int i = 0; i < 2; ++i) {
#pragma unroll
    for (int j = 0; j < 4; ++j) {
      int n0 = colBase + wc * 64 + j * 16;
      if (n0 >= N) continue;
      int col = n0 + m16;
#pragma unroll
      for (int v = 0; v < 8; ++v) {
        int rowLocal = rowBase + wr * 32 + i * 16 + v + g * 8;
        float val = acc[i][j][v];
        if (bias) val += bias[col];
        if (accumulate) val += ((float*)outp)[(size_t)rowLocal * ldo + col];
        if (relu) val = fmaxf(val, 0.f);
        int grow = rowLocal + rowOffset;
        if (outMode == OM_F32_RM) {
          ((float*)outp)[(size_t)rowLocal * ldo + col] = val;
        } else if (outMode == OM_F16_RM) {
          ((half_t*)outp)[(size_t)rowLocal * ldo + col] = (half_t)val;
        } else if (outMode == OM_F32_NCHW) {
          int b = grow / HW, hw = grow % HW;
          ((float*)outp)[((size_t)b * Cch + col) * HW + hw] = val;
        } else if (outMode == OM_F16_NCHW) {
          int b = grow / HW, hw = grow % HW;
          ((half_t*)outp)[((size_t)b * Cch + col) * HW + hw] = (half_t)val;
        } else {  // OM_F16_TN: transposed f16 store (N x M), ldo = M stride
          ((half_t*)outp)[(size_t)col * ldo + rowLocal] = (half_t)val;
        }
      }
    }
  }
}

// ---------------------------------------------------------------------------
// Support kernels
// ---------------------------------------------------------------------------
__global__ void im2col3_kernel(const float* __restrict__ src, half_t* __restrict__ dst,
                               int Bc, int C, int H, int W) {
  size_t idx = (size_t)blockIdx.x * 256 + threadIdx.x;
  size_t total = (size_t)Bc * H * W * C * 9;
  if (idx >= total) return;
  int K9 = C * 9;
  int p = (int)(idx / K9);
  int col = (int)(idx % K9);
  int c = col / 9, t = col % 9;
  int kh = t / 3, kw = t % 3;
  int hw = p % (H * W), b = p / (H * W);
  int oh = hw / W, ow = hw % W;
  int ih = oh + kh - 1, iw = ow + kw - 1;
  float v = 0.f;
  if ((unsigned)ih < (unsigned)H && (unsigned)iw < (unsigned)W)
    v = src[(((size_t)b * C + c) * H + ih) * W + iw];
  dst[idx] = (half_t)v;
}

// OIHW conv weight slice -> N x K f16 (dst[o*2304 + k], k = i*9 + kh*3 + kw)
__global__ void convw_f16_kernel(const float* __restrict__ w, half_t* __restrict__ dst,
                                 int cinTotal, int cOff) {
  size_t idx = (size_t)blockIdx.x * 256 + threadIdx.x;
  if (idx >= (size_t)2304 * 256) return;
  int o = (int)(idx / 2304);
  int k = (int)(idx % 2304);
  int i = k / 9, t = k % 9;
  dst[idx] = (half_t)w[((size_t)o * cinTotal + cOff + i) * 9 + t];
}

// f32 (K,N) row-major -> f16 (N,K) row-major
__global__ void tr_f16_kernel(const float* __restrict__ src, half_t* __restrict__ dst,
                              int K, int N) {
  size_t idx = (size_t)blockIdx.x * 256 + threadIdx.x;
  if (idx >= (size_t)K * N) return;
  int k = (int)(idx / N), n = (int)(idx % N);
  dst[(size_t)n * K + k] = (half_t)src[idx];
}

__global__ void gap_kernel(const float* __restrict__ sup, float* __restrict__ gap) {
  int idx = blockIdx.x * 256 + threadIdx.x;
  if (idx >= 20 * 256) return;
  const float* p = sup + (size_t)idx * 256;  // (s,c) -> 16*16 values
  float s = 0.f;
  for (int i = 0; i < 256; ++i) s += p[i];
  gap[idx] = s * (1.f / 256.f);
}

// center-tap projections of GAP (3x3 SAME conv on 1x1 spatial).
// gkeyT: 128 x 256 (N x K for the chl-QK GEMM), rows s>=20 zero-padded.
// gvalT: 256 x 32  (N x K for the chl-PV GEMM), cols s>=20 zero.
__global__ void gap_proj_kernel(const float* __restrict__ gap,
                                const float* __restrict__ gv_w, const float* __restrict__ gv_b,
                                const float* __restrict__ gk_w, const float* __restrict__ gk_b,
                                half_t* __restrict__ gkeyT, half_t* __restrict__ gvalT) {
  int idx = blockIdx.x * 256 + threadIdx.x;
  if (idx >= 128 * 256) return;
  int s = idx >> 8, c = idx & 255;
  float k = 0.f, v = 0.f;
  if (s < 20) {
    const float* g = gap + s * 256;
    for (int i = 0; i < 256; ++i) {
      k += gv_w[((size_t)c * 256 + i) * 9 + 4] * g[i];
      v += gk_w[((size_t)c * 256 + i) * 9 + 4] * g[i];
    }
    k += gv_b[c];
    v += gk_b[c];
  }
  gkeyT[idx] = (half_t)k;
  if (s < 32) gvalT[c * 32 + s] = (half_t)v;
}

__global__ __launch_bounds__(256) void softmax_kernel(const float* __restrict__ S,
                                                      half_t* __restrict__ P,
                                                      int L, int ld) {
  __shared__ float red[256];
  int row = blockIdx.x, tid = threadIdx.x;
  const float* s = S + (size_t)row * ld;
  float mx = -1e30f;
  for (int i = tid; i < L; i += 256) mx = fmaxf(mx, s[i]);
  red[tid] = mx; __syncthreads();
  for (int k = 128; k > 0; k >>= 1) { if (tid < k) red[tid] = fmaxf(red[tid], red[tid + k]); __syncthreads(); }
  mx = red[0]; __syncthreads();
  float sm = 0.f;
  for (int i = tid; i < L; i += 256) sm += __expf(s[i] - mx);
  red[tid] = sm; __syncthreads();
  for (int k = 128; k > 0; k >>= 1) { if (tid < k) red[tid] += red[tid + k]; __syncthreads(); }
  float inv = 1.f / red[0];
  half_t* p = P + (size_t)row * ld;
  for (int i = tid; i < ld; i += 256)
    p[i] = (i < L) ? (half_t)(__expf(s[i] - mx) * inv) : (half_t)0.f;
}

// relu + (pixel,chan) row-major -> (b, c*49+hw) channel-major f16 for the FCs
__global__ void fuse_pack_kernel(const float* __restrict__ catbuf, half_t* __restrict__ out) {
  size_t idx = (size_t)blockIdx.x * 256 + threadIdx.x;
  if (idx >= (size_t)25088 * 256) return;
  int p = (int)(idx >> 8), c = (int)(idx & 255);
  float v = fmaxf(catbuf[idx], 0.f);
  int b = p / 49, hw = p % 49;
  out[(size_t)b * 12544 + c * 49 + hw] = (half_t)v;
}

// ---------------------------------------------------------------------------
static void gemm(hipStream_t st, const half_t* A, int lda, const half_t* B, int ldb,
                 void* out, int ldo, int om, const float* bias, int acc,
                 int relu, int M, int N, int K, int rowOff, int HW, int C) {
  dim3 g((N + BN - 1) / BN, M / BM);
  wmma_gemm_kernel<<<g, 256, 0, st>>>(A, lda, B, ldb, out, ldo, om, bias, acc,
                                      relu, M, N, K, rowOff, HW, C);
}

static inline unsigned nblk(size_t n) { return (unsigned)((n + 255) / 256); }

extern "C" void kernel_launch(void* const* d_in, const int* in_sizes, int n_in,
                              void* d_out, int out_size, void* d_ws, size_t ws_size,
                              hipStream_t stream) {
  (void)in_sizes; (void)n_in; (void)out_size; (void)ws_size;
  const float* x     = (const float*)d_in[0];
  const float* sup   = (const float*)d_in[1];
  const float* qv_w  = (const float*)d_in[2];  const float* qv_b  = (const float*)d_in[3];
  const float* qk_w  = (const float*)d_in[4];  const float* qk_b  = (const float*)d_in[5];
  const float* gk_w  = (const float*)d_in[6];  const float* gk_b  = (const float*)d_in[7];
  const float* gv_w  = (const float*)d_in[8];  const float* gv_b  = (const float*)d_in[9];
  const float* sv_w  = (const float*)d_in[10]; const float* sv_b  = (const float*)d_in[11];
  const float* sk_w  = (const float*)d_in[12]; const float* sk_b  = (const float*)d_in[13];
  const float* cat_w = (const float*)d_in[14]; const float* cat_b = (const float*)d_in[15];
  const float* fc6c_w = (const float*)d_in[16]; const float* fc6c_b = (const float*)d_in[17];
  const float* fc7c_w = (const float*)d_in[18]; const float* fc7c_b = (const float*)d_in[19];
  const float* fc6r_w = (const float*)d_in[20]; const float* fc6r_b = (const float*)d_in[21];
  const float* fc7r_w = (const float*)d_in[22]; const float* fc7r_b = (const float*)d_in[23];

  constexpr int Bx = 512, Cc = 256;
  constexpr int P = 25088;            // 512*7*7 query pixels
  constexpr int Ps = 5120;            // 20*16*16 support pixels
  constexpr int KC = 2304;            // 256*9 im2col K
  constexpr int FIN = 12544, REP = 1024;
  constexpr int CHUNK = 3584, NCHUNK = 7;   // 7*3584 = 25088, 3584 % 128 == 0

  // ---- deterministic workspace carve ----
  char* wp = (char*)d_ws;
  auto take = [&](size_t bytes) {
    char* p = wp;
    wp += (bytes + 255) & ~(size_t)255;
    return p;
  };
  half_t* colbuf  = (half_t*)take((size_t)P * KC * 2);     // reused im2col buffer
  half_t* colbufS = (half_t*)take((size_t)Ps * KC * 2);
  half_t* wqvT   = (half_t*)take((size_t)256 * KC * 2);
  half_t* wqkT   = (half_t*)take((size_t)256 * KC * 2);
  half_t* wsvT   = (half_t*)take((size_t)256 * KC * 2);
  half_t* wskT   = (half_t*)take((size_t)256 * KC * 2);
  half_t* wcat0T = (half_t*)take((size_t)256 * KC * 2);
  half_t* wcat1T = (half_t*)take((size_t)256 * KC * 2);
  half_t* wcat2T = (half_t*)take((size_t)256 * KC * 2);
  half_t* wfc6cT = (half_t*)take((size_t)REP * FIN * 2);
  half_t* wfc6rT = (half_t*)take((size_t)REP * FIN * 2);
  half_t* wfc7cT = (half_t*)take((size_t)REP * REP * 2);
  half_t* wfc7rT = (half_t*)take((size_t)REP * REP * 2);
  half_t* qk16  = (half_t*)take((size_t)P * 256 * 2);
  float*  qvf   = (float*)take((size_t)P * 256 * 4);
  half_t* sk16  = (half_t*)take((size_t)Ps * 256 * 2);   // N x K for QK^T
  half_t* svT   = (half_t*)take((size_t)256 * Ps * 2);   // N x K for PV
  float*  gapb  = (float*)take((size_t)20 * 256 * 4);
  half_t* gkeyT = (half_t*)take((size_t)128 * 256 * 2);  // padded to 128 rows
  half_t* gvalT = (half_t*)take((size_t)256 * 32 * 2);
  float*  schl  = (float*)take((size_t)P * 32 * 4);
  half_t* pchl  = (half_t*)take((size_t)P * 32 * 2);
  float*  attnf  = (float*)take((size_t)P * 256 * 4);
  float*  attncf = (float*)take((size_t)P * 256 * 4);
  float*  Sbuf  = (float*)take((size_t)CHUNK * Ps * 4);
  half_t* Pbuf  = (half_t*)take((size_t)CHUNK * Ps * 2);
  float*  catbuf = (float*)take((size_t)P * 256 * 4);
  half_t* fuse16 = (half_t*)take((size_t)Bx * FIN * 2);
  half_t* h6     = (half_t*)take((size_t)Bx * REP * 2);

  // ---- pack weights (all B operands stored N x K) ----
  convw_f16_kernel<<<2304, 256, 0, stream>>>(qv_w, wqvT, 256, 0);
  convw_f16_kernel<<<2304, 256, 0, stream>>>(qk_w, wqkT, 256, 0);
  convw_f16_kernel<<<2304, 256, 0, stream>>>(sv_w, wsvT, 256, 0);
  convw_f16_kernel<<<2304, 256, 0, stream>>>(sk_w, wskT, 256, 0);
  convw_f16_kernel<<<2304, 256, 0, stream>>>(cat_w, wcat0T, 768, 0);
  convw_f16_kernel<<<2304, 256, 0, stream>>>(cat_w, wcat1T, 768, 256);
  convw_f16_kernel<<<2304, 256, 0, stream>>>(cat_w, wcat2T, 768, 512);
  tr_f16_kernel<<<nblk((size_t)FIN * REP), 256, 0, stream>>>(fc6c_w, wfc6cT, FIN, REP);
  tr_f16_kernel<<<nblk((size_t)FIN * REP), 256, 0, stream>>>(fc6r_w, wfc6rT, FIN, REP);
  tr_f16_kernel<<<nblk((size_t)REP * REP), 256, 0, stream>>>(fc7c_w, wfc7cT, REP, REP);
  tr_f16_kernel<<<nblk((size_t)REP * REP), 256, 0, stream>>>(fc7r_w, wfc7rT, REP, REP);

  // ---- GAP branch (tiny) ----
  gap_kernel<<<20, 256, 0, stream>>>(sup, gapb);
  gap_proj_kernel<<<128, 256, 0, stream>>>(gapb, gv_w, gv_b, gk_w, gk_b, gkeyT, gvalT);

  // ---- query convs (shared im2col of x) ----
  im2col3_kernel<<<nblk((size_t)P * KC), 256, 0, stream>>>(x, colbuf, Bx, Cc, 7, 7);
  gemm(stream, colbuf, KC, wqkT, KC, qk16, 256, OM_F16_RM, qk_b, 0, 0, P, 256, KC, 0, 0, 0);
  gemm(stream, colbuf, KC, wqvT, KC, qvf, 256, OM_F32_NCHW, qv_b, 0, 0, P, 256, KC, 0, 49, 256);

  // ---- support convs ----
  im2col3_kernel<<<nblk((size_t)Ps * KC), 256, 0, stream>>>(sup, colbufS, 20, Cc, 16, 16);
  gemm(stream, colbufS, KC, wskT, KC, sk16, 256, OM_F16_RM, sk_b, 0, 0, Ps, 256, KC, 0, 0, 0);
  gemm(stream, colbufS, KC, wsvT, KC, svT, Ps, OM_F16_TN, sv_b, 0, 0, Ps, 256, KC, 0, 0, 0);

  // ---- channel attention (N padded 20 -> 32, B rows padded to 128) ----
  gemm(stream, qk16, 256, gkeyT, 256, schl, 32, OM_F32_RM, nullptr, 0, 0, P, 32, 256, 0, 0, 0);
  softmax_kernel<<<P, 256, 0, stream>>>(schl, pchl, 20, 32);
  gemm(stream, pchl, 32, gvalT, 32, attncf, 256, OM_F32_NCHW, nullptr, 0, 0, P, 256, 32, 0, 49, 256);

  // ---- spatial attention, chunked over query rows ----
  for (int ch = 0; ch < NCHUNK; ++ch) {
    const half_t* Aq = qk16 + (size_t)ch * CHUNK * 256;
    gemm(stream, Aq, 256, sk16, 256, Sbuf, Ps, OM_F32_RM, nullptr, 0, 0, CHUNK, Ps, 256, 0, 0, 0);
    softmax_kernel<<<CHUNK, 256, 0, stream>>>(Sbuf, Pbuf, Ps, Ps);
    gemm(stream, Pbuf, Ps, svT, Ps, attnf, 256, OM_F32_NCHW, nullptr, 0, 0,
         CHUNK, 256, Ps, ch * CHUNK, 49, 256);
  }

  // ---- fuse conv over concat = sum of three 256-channel convs ----
  im2col3_kernel<<<nblk((size_t)P * KC), 256, 0, stream>>>(qvf, colbuf, Bx, Cc, 7, 7);
  gemm(stream, colbuf, KC, wcat0T, KC, catbuf, 256, OM_F32_RM, cat_b, 0, 0, P, 256, KC, 0, 0, 0);
  im2col3_kernel<<<nblk((size_t)P * KC), 256, 0, stream>>>(attnf, colbuf, Bx, Cc, 7, 7);
  gemm(stream, colbuf, KC, wcat1T, KC, catbuf, 256, OM_F32_RM, nullptr, 1, 0, P, 256, KC, 0, 0, 0);
  im2col3_kernel<<<nblk((size_t)P * KC), 256, 0, stream>>>(attncf, colbuf, Bx, Cc, 7, 7);
  gemm(stream, colbuf, KC, wcat2T, KC, catbuf, 256, OM_F32_RM, nullptr, 1, 0, P, 256, KC, 0, 0, 0);
  fuse_pack_kernel<<<nblk((size_t)P * 256), 256, 0, stream>>>(catbuf, fuse16);

  // ---- FC heads: xc then xr (d_out = [xc | xr]) ----
  float* out_c = (float*)d_out;
  float* out_r = (float*)d_out + (size_t)Bx * REP;
  gemm(stream, fuse16, FIN, wfc6cT, FIN, h6, REP, OM_F16_RM, fc6c_b, 0, 1, Bx, REP, FIN, 0, 0, 0);
  gemm(stream, h6, REP, wfc7cT, REP, out_c, REP, OM_F32_RM, fc7c_b, 0, 1, Bx, REP, REP, 0, 0, 0);
  gemm(stream, fuse16, FIN, wfc6rT, FIN, h6, REP, OM_F16_RM, fc6r_b, 0, 1, Bx, REP, FIN, 0, 0, 0);
  gemm(stream, h6, REP, wfc7rT, REP, out_r, REP, OM_F32_RM, fc7r_b, 0, 1, Bx, REP, REP, 0, 0, 0);
}
